// RPNBoxSelector_25975962206920
// MI455X (gfx1250) — compile-verified, matrix operand order checked by reading, not run on hardware
//
#include <hip/hip_runtime.h>
#include <hip/hip_bf16.h>

typedef unsigned int u32;
typedef __attribute__((ext_vector_type(16))) _Float16 v16h;
typedef __attribute__((ext_vector_type(8)))  float    v8f;

// Problem constants (match reference)
constexpr int N   = 8;
constexpr int A   = 15;
constexpr int H   = 160;
constexpr int W   = 160;
constexpr int HW  = H * W;          // 25600
constexpr int HWA = HW * A;         // 384000
constexpr int K   = 3000;           // PRE_NMS_TOP_N
constexpr int KP  = 3008;           // padded to 16 (WMMA tile) and 32 (mask words)
constexpr int WORDS = KP / 32;      // 94
constexpr int POST  = 1000;
constexpr float NMS_T = 0.7f;
constexpr float BBOX_CLIP = 4.135166556742356f;  // log(1000/16)

// ---------------- workspace layout (bytes, 256-aligned) ----------------
constexpr size_t OFF_KEYS   = 0;                               // N*HWA u32      = 12,288,000
constexpr size_t OFF_HIST   = OFF_KEYS + (size_t)N*HWA*4;      // 4*N*256 u32    = 32,768
constexpr size_t OFF_STATE  = OFF_HIST + 4*N*256*4;            // 5*N u32 (pad 256)
constexpr size_t OFF_TKEY   = OFF_STATE + 256;                 // N*KP u32
constexpr size_t OFF_TIDX   = OFF_TKEY + (size_t)N*KP*4;       // N*KP u32
constexpr size_t OFF_BOXES  = OFF_TIDX + (size_t)N*KP*4;       // N*KP float4
constexpr size_t OFF_CULL   = OFF_BOXES + (size_t)N*KP*16;     // N*KP float4 (cx/16, cy/16, rad/16, area)
constexpr size_t OFF_SCORE  = OFF_CULL + (size_t)N*KP*16;      // N*KP float
constexpr size_t OFF_MASK   = OFF_SCORE + (size_t)N*KP*4;      // N*KP*WORDS u32 = 9,048,064

// state array slots
#define ST_PREFIX 0
#define ST_NEED   (N)
#define ST_GCNT   (2*N)
#define ST_CGT    (3*N)
#define ST_CEQ    (4*N)

// ---------------------------------------------------------------------
__global__ void zero32_kernel(u32* p, int nWords) {
    int i = blockIdx.x * blockDim.x + threadIdx.x;
    if (i < nWords) p[i] = 0u;
}

__global__ void init_state_kernel(u32* state) {
    int t = threadIdx.x;
    if (t < N) {
        state[ST_PREFIX + t] = 0u;
        state[ST_NEED   + t] = (u32)K;
        state[ST_GCNT   + t] = 0u;
        state[ST_CGT    + t] = 0u;
        state[ST_CEQ    + t] = 0u;
    }
}

// NCHW logits -> HWA-ordered monotone sort keys
__global__ void keys_kernel(const float* __restrict__ obj, u32* __restrict__ keys) {
    int n   = blockIdx.y;
    int hwa = blockIdx.x * blockDim.x + threadIdx.x;   // grid sized exactly
    int a   = hwa % A;
    int hw  = hwa / A;
    float x = obj[((size_t)n * A + a) * HW + hw];
    u32 u = __float_as_uint(x);
    u32 key = (u & 0x80000000u) ? ~u : (u | 0x80000000u);   // bigger float -> bigger key
    keys[(size_t)n * HWA + hwa] = key;
}

// MSB-first 8-bit radix-select histogram pass p (p = 0..3)
__global__ void hist_kernel(const u32* __restrict__ keys, const u32* __restrict__ state,
                            u32* __restrict__ hist, int p) {
    __shared__ u32 lh[256];
    int tid = threadIdx.x;
    lh[tid] = 0;
    __syncthreads();
    int n = blockIdx.y;
    int i = blockIdx.x * blockDim.x + tid;
    u32 key = keys[(size_t)n * HWA + i];
    bool match = true;
    if (p > 0) {
        u32 m = 0xFFFFFFFFu << (32 - 8 * p);
        match = ((key & m) == (state[ST_PREFIX + n] & m));
    }
    if (match) {
        int shift = 24 - 8 * p;
        atomicAdd(&lh[(key >> shift) & 0xFFu], 1u);
    }
    __syncthreads();
    if (lh[tid]) atomicAdd(&hist[((size_t)p * N + n) * 256 + tid], lh[tid]);
}

__global__ void select_kernel(const u32* __restrict__ hist, u32* state, int p) {
    int n = blockIdx.x;
    if (threadIdx.x != 0) return;
    const u32* h = hist + ((size_t)p * N + n) * 256;
    u32 need = state[ST_NEED + n];
    u32 cum = 0;
    int b = 255;
    for (; b >= 0; --b) {
        u32 c = h[b];
        if (cum + c >= need) break;
        cum += c;
    }
    if (b < 0) b = 0;
    int shift = 24 - 8 * p;
    state[ST_PREFIX + n] |= ((u32)b) << shift;
    state[ST_GCNT   + n] += cum;
    state[ST_NEED   + n]  = need - cum;
}

// scatter items above/equal threshold into [0, K) slots (unsorted)
__global__ void compact_kernel(const u32* __restrict__ keys, u32* state,
                               u32* __restrict__ tkey, u32* __restrict__ tidx) {
    int n = blockIdx.y;
    int i = blockIdx.x * blockDim.x + threadIdx.x;
    u32 key = keys[(size_t)n * HWA + i];
    u32 T = state[ST_PREFIX + n];
    if (key > T) {
        u32 pos = atomicAdd(&state[ST_CGT + n], 1u);
        tkey[(size_t)n * KP + pos] = key;
        tidx[(size_t)n * KP + pos] = (u32)i;
    } else if (key == T) {
        u32 G = state[ST_GCNT + n];
        u32 e = atomicAdd(&state[ST_CEQ + n], 1u);
        u32 pos = G + e;
        if (pos < (u32)K) {
            tkey[(size_t)n * KP + pos] = key;
            tidx[(size_t)n * KP + pos] = (u32)i;
        }
    }
}

// in-LDS bitonic sort of the 3000 survivors (pad to 4096), key desc / idx asc
__global__ __launch_bounds__(1024) void sort_kernel(u32* tkey, u32* tidx) {
    __shared__ u32 sk[4096];
    __shared__ u32 si[4096];
    int n = blockIdx.x, tid = threadIdx.x;
    for (int e = tid; e < 4096; e += 1024) {
        if (e < K) { sk[e] = tkey[(size_t)n * KP + e]; si[e] = tidx[(size_t)n * KP + e]; }
        else       { sk[e] = 0u;                       si[e] = 0xFFFFFFFFu; }
    }
    for (int k = 2; k <= 4096; k <<= 1) {
        for (int j = k >> 1; j > 0; j >>= 1) {
            __syncthreads();
            for (int e = tid; e < 4096; e += 1024) {
                int x = e ^ j;
                if (x > e) {
                    u32 ka = sk[e], kb = sk[x], ia = si[e], ib = si[x];
                    bool before = (ka > kb) || (ka == kb && ia < ib);  // "comes earlier"
                    bool asc = ((e & k) == 0);
                    if (before != asc) { sk[e] = kb; sk[x] = ka; si[e] = ib; si[x] = ia; }
                }
            }
        }
    }
    __syncthreads();
    for (int e = tid; e < K; e += 1024) {
        tkey[(size_t)n * KP + e] = sk[e];
        tidx[(size_t)n * KP + e] = si[e];
    }
}

// decode boxes at sorted indices + precompute cull features + sigmoid scores
__global__ void decode_kernel(const float* __restrict__ anchors,
                              const float* __restrict__ box_reg,
                              const u32* __restrict__ tkey, const u32* __restrict__ tidx,
                              float4* __restrict__ boxes, float4* __restrict__ cull,
                              float* __restrict__ scores) {
    int gid = blockIdx.x * blockDim.x + threadIdx.x;
    if (gid >= N * KP) return;
    int n = gid / KP, k = gid % KP;
    if (k >= K) {  // pad rows: never overlap anything (rad = -inf-ish)
        boxes[gid]  = make_float4(0.f, 0.f, 0.f, 0.f);
        cull[gid]   = make_float4(0.f, 0.f, -1e30f, 1.f);
        scores[gid] = 0.f;
        return;
    }
    u32 idx = tidx[(size_t)n * KP + k];
    int a  = (int)(idx % A);
    int hw = (int)(idx / A);
    const float* anc = anchors + ((size_t)n * HWA + idx) * 4;
    float a0 = anc[0], a1 = anc[1], a2 = anc[2], a3 = anc[3];
    float wv = a2 - a0 + 1.f, hv = a3 - a1 + 1.f;
    float cx = a0 + 0.5f * wv, cy = a1 + 0.5f * hv;
    size_t rbase = (size_t)n * (A * 4) * HW + (size_t)hw;
    float dx = box_reg[rbase + (size_t)(a * 4 + 0) * HW];
    float dy = box_reg[rbase + (size_t)(a * 4 + 1) * HW];
    float dw = fminf(box_reg[rbase + (size_t)(a * 4 + 2) * HW], BBOX_CLIP);
    float dh = fminf(box_reg[rbase + (size_t)(a * 4 + 3) * HW], BBOX_CLIP);
    float pcx = dx * wv + cx, pcy = dy * hv + cy;
    float pw = __expf(dw) * wv, ph = __expf(dh) * hv;
    float x1 = pcx - 0.5f * pw, y1 = pcy - 0.5f * ph;
    float x2 = pcx + 0.5f * pw - 1.f, y2 = pcy + 0.5f * ph - 1.f;
    boxes[gid] = make_float4(x1, y1, x2, y2);
    float area = pw * ph;                               // (x2-x1+1)*(y2-y1+1)
    const float S = 1.f / 16.f;                         // keep centers in f16 sweet spot
    float rad = 0.5f * sqrtf(pw * pw + ph * ph) * S + 8.f;  // + margin for f16 rounding
    cull[gid] = make_float4(pcx * S, pcy * S, rad, area);
    // score = sigmoid(logit); recover logit from the monotone key
    u32 key = tkey[(size_t)n * KP + k];
    u32 u = (key & 0x80000000u) ? (key & 0x7FFFFFFFu) : ~key;
    float logit = __uint_as_float(u);
    scores[gid] = 1.f / (1.f + __expf(-logit));
}

// 16x16 tile of the suppression bitmask per wave32.
// WMMA computes pairwise center dot products (rank-2 GEMM) for a distance cull;
// surviving pairs get a division-free exact IoU test on the VALU:
//   inter/(Ai+Aj-inter) > T  <=>  inter*(1+T) > T*(Ai+Aj)
__global__ __launch_bounds__(32) void mask_kernel(const float4* __restrict__ boxes,
                                                  const float4* __restrict__ cull,
                                                  u32* __restrict__ mask) {
    int n  = blockIdx.z;
    int tr = blockIdx.y * 16;   // tile row base
    int tc = blockIdx.x * 16;   // tile col base
    __shared__ float4 rB[16], cB[16], rC[16], cC[16];
    int lane = threadIdx.x;
    if (lane < 16) {
        rB[lane] = boxes[(size_t)n * KP + tr + lane];
        rC[lane] = cull [(size_t)n * KP + tr + lane];
    } else {
        int l = lane - 16;
        cB[l] = boxes[(size_t)n * KP + tc + l];
        cC[l] = cull [(size_t)n * KP + tc + l];
    }
    __syncthreads();

    // A: rows' (cx,cy) in K=0,1 ; B: cols' (cx,cy) in K=0,1 ; rest zero.
    v16h av = {};
    v16h bv = {};
    if (lane < 16) {
        av[0] = (_Float16)rC[lane].x;  av[1] = (_Float16)rC[lane].y;
        bv[0] = (_Float16)cC[lane].x;  bv[1] = (_Float16)cC[lane].y;
    }
    v8f d = {};
    d = __builtin_amdgcn_wmma_f32_16x16x32_f16(false, av, false, bv,
                                               (short)0, d, false, false);

    int col   = (lane < 16) ? lane : (lane - 16);
    int rbase = (lane < 16) ? 0 : 8;
    float4 cb = cB[col];
    float4 cc = cC[col];
    float sj  = cc.x * cc.x + cc.y * cc.y;
    int wordIdx = tc >> 5;
    int sh      = tc & 31;     // 0 or 16

#pragma unroll
    for (int r = 0; r < 8; ++r) {
        int i = rbase + r;
        float4 rb = rB[i];
        float4 rc = rC[i];
        float si = rc.x * rc.x + rc.y * rc.y;
        float dist2 = si + sj - 2.0f * d[r];
        float rsum  = rc.z + cc.z;
        bool bit = false;
        int gi = tr + i, gj = tc + col;
        if (rsum > 0.f && dist2 <= rsum * rsum && gi != gj) {
            float xx1 = fmaxf(rb.x, cb.x), yy1 = fmaxf(rb.y, cb.y);
            float xx2 = fminf(rb.z, cb.z), yy2 = fminf(rb.w, cb.w);
            float iw = fmaxf(xx2 - xx1 + 1.f, 0.f);
            float ih = fmaxf(yy2 - yy1 + 1.f, 0.f);
            float inter = iw * ih;
            // division-free IoU threshold test (denominator always positive)
            bit = inter * (1.f + NMS_T) > NMS_T * (rc.w + cc.w);
        }
        unsigned v = __builtin_amdgcn_ballot_w32(bit);  // low16: rows 0-7, hi16: rows 8-15
        if (lane == r) {
            atomicOr(&mask[((size_t)n * KP + tr + r    ) * WORDS + wordIdx], (v & 0xFFFFu) << sh);
            atomicOr(&mask[((size_t)n * KP + tr + r + 8) * WORDS + wordIdx], (v >> 16)     << sh);
        }
    }
}

// serial greedy scan using the bitmask; one block per image
__global__ __launch_bounds__(128) void scan_kernel(const float4* __restrict__ boxes,
                                                   const float* __restrict__ scores,
                                                   const u32* __restrict__ mask,
                                                   float* __restrict__ out) {
    int n = blockIdx.x, tid = threadIdx.x;
    __shared__ u32 removed[WORDS];
    __shared__ int keptCount, keepFlag, stop;
    for (int wd = tid; wd < WORDS; wd += blockDim.x) removed[wd] = 0u;
    if (tid == 0) { keptCount = 0; stop = 0; }
    for (int i = 0; i < K; ++i) {
        __syncthreads();
        if (stop) break;
        if (tid == 0) {
            keepFlag = 0;
            if (!((removed[i >> 5] >> (i & 31)) & 1u)) {
                keepFlag = 1;
                int slot = keptCount++;
                float4 bx = boxes[(size_t)n * KP + i];
                float sc  = scores[(size_t)n * KP + i];
                float* o = out + ((size_t)n * POST + slot) * 5;
                o[0] = bx.x; o[1] = bx.y; o[2] = bx.z; o[3] = bx.w; o[4] = sc;
                if (keptCount >= POST) stop = 1;
            }
        }
        __syncthreads();
        if (keepFlag) {
            const u32* mrow = mask + ((size_t)n * KP + i) * WORDS;
            for (int wd = tid; wd < WORDS; wd += blockDim.x) removed[wd] |= mrow[wd];
        }
    }
}

// ---------------------------------------------------------------------
extern "C" void kernel_launch(void* const* d_in, const int* in_sizes, int n_in,
                              void* d_out, int out_size, void* d_ws, size_t ws_size,
                              hipStream_t stream) {
    const float* objectness = (const float*)d_in[0];   // [N, A, H, W]
    const float* box_reg    = (const float*)d_in[1];   // [N, A*4, H, W]
    const float* anchors    = (const float*)d_in[2];   // [N, HWA, 4]
    float* out = (float*)d_out;                        // [N, POST, 5]

    char* ws = (char*)d_ws;
    u32*    keys   = (u32*)   (ws + OFF_KEYS);
    u32*    hist   = (u32*)   (ws + OFF_HIST);
    u32*    state  = (u32*)   (ws + OFF_STATE);
    u32*    tkey   = (u32*)   (ws + OFF_TKEY);
    u32*    tidx   = (u32*)   (ws + OFF_TIDX);
    float4* boxes  = (float4*)(ws + OFF_BOXES);
    float4* cull   = (float4*)(ws + OFF_CULL);
    float*  scores = (float*) (ws + OFF_SCORE);
    u32*    mask   = (u32*)   (ws + OFF_MASK);

    // zero histograms, suppression bitmask, and output
    {
        int hw_ = 4 * N * 256;
        zero32_kernel<<<(hw_ + 255) / 256, 256, 0, stream>>>(hist, hw_);
        int mw_ = N * KP * WORDS;
        zero32_kernel<<<(mw_ + 255) / 256, 256, 0, stream>>>(mask, mw_);
        zero32_kernel<<<(out_size + 255) / 256, 256, 0, stream>>>((u32*)out, out_size);
    }
    init_state_kernel<<<1, 64, 0, stream>>>(state);

    // sortable keys in HWA order
    keys_kernel<<<dim3(HWA / 256, N), 256, 0, stream>>>(objectness, keys);

    // 4-pass exact radix select of the 3000th key per image
    for (int p = 0; p < 4; ++p) {
        hist_kernel<<<dim3(HWA / 256, N), 256, 0, stream>>>(keys, state, hist, p);
        select_kernel<<<N, 32, 0, stream>>>(hist, state, p);
    }
    compact_kernel<<<dim3(HWA / 256, N), 256, 0, stream>>>(keys, state, tkey, tidx);
    sort_kernel<<<N, 1024, 0, stream>>>(tkey, tidx);

    // decode + cull features + scores
    decode_kernel<<<(N * KP + 255) / 256, 256, 0, stream>>>(anchors, box_reg, tkey, tidx,
                                                            boxes, cull, scores);
    // WMMA-culled suppression matrix: (188 x 188 tiles) x 8 images, one wave32 each
    mask_kernel<<<dim3(KP / 16, KP / 16, N), 32, 0, stream>>>(boxes, cull, mask);

    // greedy NMS scan + compaction
    scan_kernel<<<N, 128, 0, stream>>>(boxes, scores, mask, out);
}
